// QuantizedLinear_12068858102138
// MI455X (gfx1250) — compile-verified, matrix operand order checked by reading, not run on hardware
//
#include <hip/hip_runtime.h>

typedef __attribute__((ext_vector_type(16))) __bf16 v16bf;
typedef __attribute__((ext_vector_type(8)))  float  v8f;
typedef __attribute__((ext_vector_type(4)))  float  v4f;
typedef __attribute__((ext_vector_type(4)))  int    v4i;
typedef __attribute__((ext_vector_type(4)))  unsigned int v4u;
// exact parameter type of __builtin_amdgcn_global_load_async_to_lds_b128
typedef int async_v4i __attribute__((vector_size(4 * sizeof(int))));

#define M_TOT 8192
#define N_TOT 4096
#define K_TOT 4096
#define BM 128
#define BN 128
#define BK 128
#define ROWU 68   // uints per LDS row: 128 bf16 (64 uints) + 8 bf16 pad (4 uints) = 272B
#define KU (K_TOT / 2)
#define NCHUNK (K_TOT / BK)

#if __has_builtin(__builtin_amdgcn_global_load_async_to_lds_b128) && \
    __has_builtin(__builtin_amdgcn_s_wait_asynccnt)
#define HAVE_ASYNC 1
#else
#define HAVE_ASYNC 0
#endif

__device__ __forceinline__ unsigned int pack2_bf16(float a, float b) {
  unsigned int ua = __float_as_uint(a);
  unsigned int ub = __float_as_uint(b);
  ua = (ua + 0x7FFFu + ((ua >> 16) & 1u)) >> 16;   // round-to-nearest-even
  ub = (ub + 0x7FFFu + ((ub >> 16) & 1u)) >> 16;
  return (ub << 16) | (ua & 0xFFFFu);
}

// ---- prepass 1: x f32 -> bf16 (8 elements / thread) ----
__global__ __launch_bounds__(256) void cvt_x_kernel(const float* __restrict__ x,
                                                    unsigned int* __restrict__ xb) {
  size_t base = ((size_t)blockIdx.x * 256 + threadIdx.x) * 8;
  v4f a = *(const v4f*)(x + base);
  v4f b = *(const v4f*)(x + base + 4);
  v4u p;
  p.x = pack2_bf16(a.x, a.y);
  p.y = pack2_bf16(a.z, a.w);
  p.z = pack2_bf16(b.x, b.y);
  p.w = pack2_bf16(b.z, b.w);
  *(v4u*)(xb + base / 2) = p;
}

// ---- prepass 2: weight int32 -> dequant -> bf16 (8 elements / thread) ----
__global__ __launch_bounds__(256) void cvt_w_kernel(const int* __restrict__ qw,
                                                    unsigned int* __restrict__ wb,
                                                    const int* __restrict__ zpp,
                                                    const float* __restrict__ scp) {
  const int   zp = zpp[0];
  const float sc = scp[0];
  size_t base = ((size_t)blockIdx.x * 256 + threadIdx.x) * 8;
  v4i a = *(const v4i*)(qw + base);
  v4i b = *(const v4i*)(qw + base + 4);
  v4u p;
  p.x = pack2_bf16((float)(a.x - zp) * sc, (float)(a.y - zp) * sc);
  p.y = pack2_bf16((float)(a.z - zp) * sc, (float)(a.w - zp) * sc);
  p.z = pack2_bf16((float)(b.x - zp) * sc, (float)(b.y - zp) * sc);
  p.w = pack2_bf16((float)(b.z - zp) * sc, (float)(b.w - zp) * sc);
  *(v4u*)(wb + base / 2) = p;
}

// ---- main GEMM: bf16 x bf16 -> f32, 128x128 tile, BK=128, double-buffered LDS ----
__global__ __launch_bounds__(256) void qlinear_wmma_kernel(
    const unsigned int* __restrict__ xb,   // [M_TOT][KU] bf16 pairs
    const unsigned int* __restrict__ wb,   // [N_TOT][KU] bf16 pairs
    const float* __restrict__ bias, float* __restrict__ out) {
  __shared__ unsigned int ldsA[2][BM * ROWU];
  __shared__ unsigned int ldsB[2][BN * ROWU];

  const int tid   = threadIdx.x;
  const int lane  = tid & 31;
  const int wid   = tid >> 5;
  const int lmod  = lane & 15;
  const int khalf = lane >> 4;
  const int wm = wid >> 2;       // 0..1  -> 64-row strip of M
  const int wn = wid & 3;        // 0..3  -> 32-col strip of N

  const int n0 = blockIdx.x * BN;
  const int m0 = blockIdx.y * BM;

  v8f acc[4][2] = {};

  // ---- cooperative copy of one 128-wide K chunk into LDS buffer `bufi` ----
  auto issue_copy = [&](int k0u, int bufi) {
#pragma unroll
    for (int i = 0; i < 8; ++i) {
      int idx = i * 256 + tid;        // 0..2047
      int row = idx >> 4;             // 0..127
      int cv  = idx & 15;             // 16B chunk within 256B row
      const unsigned int* gA = xb + (size_t)(m0 + row) * KU + k0u + cv * 4;
      const unsigned int* gB = wb + (size_t)(n0 + row) * KU + k0u + cv * 4;
      unsigned int* lA = &ldsA[bufi][row * ROWU + cv * 4];
      unsigned int* lB = &ldsB[bufi][row * ROWU + cv * 4];
#if HAVE_ASYNC
      __builtin_amdgcn_global_load_async_to_lds_b128(
          (async_v4i*)gA, (async_v4i*)lA, 0, 0);
      __builtin_amdgcn_global_load_async_to_lds_b128(
          (async_v4i*)gB, (async_v4i*)lB, 0, 0);
#else
      v4u a = *(const v4u*)gA;
      *(v4u*)lA = a;
      v4u b = *(const v4u*)gB;
      *(v4u*)lB = b;
#endif
    }
  };

  // ---- fragment loads for one K=32 substep ----
  auto load_frags = [&](const unsigned int* bA, const unsigned int* bB, int kk,
                        v16bf* fa, v16bf* fb) {
    const int kOffU = kk * 16 + khalf * 8;
#pragma unroll
    for (int i = 0; i < 4; ++i) {
      int row = wm * 64 + i * 16 + lmod;
      union { v4u q[2]; v16bf v; } u;
      const v4u* p = (const v4u*)(bA + row * ROWU + kOffU);
      u.q[0] = p[0]; u.q[1] = p[1];
      fa[i] = u.v;
    }
#pragma unroll
    for (int j = 0; j < 2; ++j) {
      int col = wn * 32 + j * 16 + lmod;
      union { v4u q[2]; v16bf v; } u;
      const v4u* p = (const v4u*)(bB + col * ROWU + kOffU);
      u.q[0] = p[0]; u.q[1] = p[1];
      fb[j] = u.v;
    }
  };

  issue_copy(0, 0);
#if HAVE_ASYNC
  __builtin_amdgcn_s_wait_asynccnt(0);
#endif
  __syncthreads();

  for (int c = 0; c < NCHUNK; ++c) {
    const int p = c & 1;
    if (c + 1 < NCHUNK) issue_copy((c + 1) * (BK / 2), (c + 1) & 1);

    const unsigned int* bA = ldsA[p];
    const unsigned int* bB = ldsB[p];

    // software-pipelined substeps: load frags(kk+1) before WMMAs(kk)
    v16bf fa[2][4], fb[2][2];
    load_frags(bA, bB, 0, fa[0], fb[0]);
#pragma unroll
    for (int kk = 0; kk < 4; ++kk) {
      const int cur = kk & 1;
      const int nxt = cur ^ 1;
      if (kk < 3) load_frags(bA, bB, kk + 1, fa[nxt], fb[nxt]);
#pragma unroll
      for (int i = 0; i < 4; ++i)
#pragma unroll
        for (int j = 0; j < 2; ++j)
          acc[i][j] = __builtin_amdgcn_wmma_f32_16x16x32_bf16(
              false, fa[cur][i], false, fb[cur][j], (short)0, acc[i][j], false, false);
    }

#if HAVE_ASYNC
    __builtin_amdgcn_s_wait_asynccnt(0);
#endif
    __syncthreads();
  }

  // ---- epilogue: add bias, store f32 ----
#pragma unroll
  for (int i = 0; i < 4; ++i) {
#pragma unroll
    for (int j = 0; j < 2; ++j) {
      int n = n0 + wn * 32 + j * 16 + lmod;
      float bv = bias[n];
      int mbase = m0 + wm * 64 + i * 16 + khalf * 8;
      v8f c = acc[i][j];
#pragma unroll
      for (int e = 0; e < 8; ++e) {
        out[(size_t)(mbase + e) * N_TOT + n] = c[e] + bv;
      }
    }
  }
}

extern "C" void kernel_launch(void* const* d_in, const int* in_sizes, int n_in,
                              void* d_out, int out_size, void* d_ws, size_t ws_size,
                              hipStream_t stream) {
  const float* x    = (const float*)d_in[0];
  const int*   qw   = (const int*)d_in[1];
  const int*   zp   = (const int*)d_in[2];
  const float* sc   = (const float*)d_in[3];
  const float* bias = (const float*)d_in[4];
  float* out = (float*)d_out;

  // workspace layout: x_bf16 (64 MiB) | w_bf16 (32 MiB)
  unsigned int* xb = (unsigned int*)d_ws;
  unsigned int* wb = (unsigned int*)((char*)d_ws + (size_t)M_TOT * K_TOT * 2);

  // prepass conversions (each thread handles 8 elements)
  cvt_x_kernel<<<(M_TOT * (size_t)K_TOT) / (256 * 8), 256, 0, stream>>>(x, xb);
  cvt_w_kernel<<<(N_TOT * (size_t)K_TOT) / (256 * 8), 256, 0, stream>>>(qw, wb, zp, sc);

  // main GEMM
  dim3 grid(N_TOT / BN, M_TOT / BM);   // (32, 64)
  qlinear_wmma_kernel<<<grid, 256, 0, stream>>>(xb, wb, bias, out);
}
